// FermiNet_76596446757266
// MI455X (gfx1250) — matrix operand order repaired and checked by reading, not compiled, not run on hardware
//
#include <hip/hip_runtime.h>

typedef __attribute__((ext_vector_type(16))) __bf16 v16bf;
typedef __attribute__((ext_vector_type(8)))  float  v8f;

#define B_WALK 4096
#define N_EL   16
#define D1     256
#define D2     32
#define FLEN   832
#define KSTEPS 26   // FLEN / 32

union Frag { v16bf v; uint4 q[2]; };

// ---- fp32x2 -> packed bf16x2 (RNE via HW cvt when available) --------------
__device__ __forceinline__ unsigned int pk2(float lo, float hi) {
#if __has_builtin(__builtin_amdgcn_cvt_pk_bf16_f32)
    auto p = __builtin_amdgcn_cvt_pk_bf16_f32(lo, hi);   // v_cvt_pk_bf16_f32
    return __builtin_bit_cast(unsigned int, p);
#else
    // round-to-nearest (ties away) + byte-splice of the two high halves
    unsigned int ul = __float_as_uint(lo) + 0x8000u;
    unsigned int uh = __float_as_uint(hi) + 0x8000u;
    return __builtin_amdgcn_perm(uh, ul, 0x07060302u);   // {uh[3],uh[2],ul[3],ul[2]}
#endif
}

// ---------------------------------------------------------------------------
// Spin-resolved mean pools: g_up/g_dn (B,D1), dg_up/dg_dn (B,N,D2)
// ---------------------------------------------------------------------------
__global__ void __launch_bounds__(256)
fermi_pool_g(const float* __restrict__ sh, float* __restrict__ gup,
             float* __restrict__ gdn, const int* __restrict__ nupp) {
    int idx = blockIdx.x * 256 + threadIdx.x;        // b*D1 + d
    if (idx >= B_WALK * D1) return;
    int b = idx >> 8;
    int d = idx & (D1 - 1);
    int nup = *nupp;
    const float* base = sh + (size_t)b * N_EL * D1 + d;
    float su = 0.f, sd = 0.f;
    #pragma unroll
    for (int i = 0; i < N_EL; ++i) {
        float v = base[i * D1];
        if (i < nup) su += v; else sd += v;
    }
    gup[idx] = su / (float)nup;
    gdn[idx] = sd / (float)(N_EL - nup);
}

__global__ void __launch_bounds__(256)
fermi_pool_dg(const float* __restrict__ dh, float* __restrict__ dgu,
              float* __restrict__ dgd, const int* __restrict__ nupp) {
    int idx = blockIdx.x * 256 + threadIdx.x;        // (b*N+i)*D2 + e
    if (idx >= B_WALK * N_EL * D2) return;
    int e  = idx & (D2 - 1);
    int bi = idx >> 5;
    int nup = *nupp;
    const float* base = dh + (size_t)bi * N_EL * D2 + e;
    float su = 0.f, sd = 0.f;
    #pragma unroll
    for (int j = 0; j < N_EL; ++j) {
        float v = base[j * D2];
        if (j < nup) su += v; else sd += v;
    }
    dgu[idx] = su / (float)nup;
    dgd[idx] = sd / (float)(N_EL - nup);
}

// ---------------------------------------------------------------------------
// v (N, FLEN, D1) fp32  ->  vt (N, D1, FLEN) bf16, packed pair store along k
// ---------------------------------------------------------------------------
__global__ void __launch_bounds__(256)
fermi_prep_vt(const float* __restrict__ v, unsigned int* __restrict__ vt) {
    int idx = blockIdx.x * 256 + threadIdx.x;        // i*(FLEN/2)*D1 + k2*D1 + n
    if (idx >= N_EL * (FLEN / 2) * D1) return;
    int n   = idx & (D1 - 1);
    int ik  = idx >> 8;
    int k2  = ik % (FLEN / 2);
    int i   = ik / (FLEN / 2);
    const float* src = v + ((size_t)i * FLEN + 2 * k2) * D1 + n;
    // vt row (i,n) is k-contiguous; uint index = (row*FLEN + 2*k2)/2
    vt[(size_t)(i * D1 + n) * (FLEN / 2) + k2] = pk2(src[0], src[D1]);
}

// ---------------------------------------------------------------------------
// Single stream: out[b,i,:] = tanh(f[b,i,:] @ v[i] + b[i]) + single_h[b,i,:]
// Block = 256 thr (8 waves), tile M=128 (batch) x N=128, K-loop of 26x32 bf16.
// ---------------------------------------------------------------------------
__global__ void __launch_bounds__(256)
fermi_single_gemm(const float* __restrict__ sh,
                  const float* __restrict__ gup, const float* __restrict__ gdn,
                  const float* __restrict__ dgu, const float* __restrict__ dgd,
                  const unsigned short* __restrict__ vt,
                  const float* __restrict__ bvec,
                  float* __restrict__ out)
{
    __shared__ unsigned int a_lds[2][128 * 20];      // 128 rows x 32 bf16, pitch 80B

    const int i      = blockIdx.z;
    const int mBase  = blockIdx.x * 128;
    const int nBase  = blockIdx.y * 128;
    const int tid    = threadIdx.x;
    const int lane   = tid & 31;
    const int wv     = tid >> 5;
    const int waveM  = wv & 3;                       // 4 waves along M
    const int waveN  = wv >> 2;                      // 2 waves along N
    const int laneLo = lane & 15;
    const int laneHi = lane >> 4;

    // A staging: 2 threads per LDS row, 16 fp32 each -> 16 bf16
    const int r    = tid >> 1;
    const int half = tid & 1;
    const int bRow = mBase + r;

    const float* srcS = sh  + (size_t)(bRow * N_EL + i) * D1 + half * 16;
    const float* srcU = gup + (size_t)bRow * D1 + half * 16;
    const float* srcD = gdn + (size_t)bRow * D1 + half * 16;
    const float* srcP = dgu + (size_t)(bRow * N_EL + i) * D2 + half * 16;
    const float* srcQ = dgd + (size_t)(bRow * N_EL + i) * D2 + half * 16;

    auto srcPtr = [&](int ks) -> const float* {      // 32-wide K chunk -> concat region
        if (ks < 8)  return srcS + ks * 32;
        if (ks < 16) return srcU + (ks - 8) * 32;
        if (ks < 24) return srcD + (ks - 16) * 32;
        return (ks == 24) ? srcP : srcQ;
    };

    // B fragment base pointers (bf16 vt rows are k-contiguous; L2-resident)
    const uint4* bp[4];
    #pragma unroll
    for (int sn = 0; sn < 4; ++sn) {
        int n = nBase + waveN * 64 + sn * 16 + laneLo;
        bp[sn] = (const uint4*)(vt + (size_t)(i * D1 + n) * FLEN) + laneHi;
    }

    const v8f vzero = {0.f, 0.f, 0.f, 0.f, 0.f, 0.f, 0.f, 0.f};
    v8f acc[2][4];
    #pragma unroll
    for (int sm = 0; sm < 2; ++sm)
        #pragma unroll
        for (int sn = 0; sn < 4; ++sn)
            acc[sm][sn] = vzero;

    // prologue: fill buffer 0
    {
        const float4* s4 = (const float4*)srcPtr(0);
        float4 t0 = s4[0], t1 = s4[1], t2 = s4[2], t3 = s4[3];
        unsigned int* dst = &a_lds[0][r * 20 + half * 8];
        ((uint4*)dst)[0] = make_uint4(pk2(t0.x,t0.y), pk2(t0.z,t0.w), pk2(t1.x,t1.y), pk2(t1.z,t1.w));
        ((uint4*)dst)[1] = make_uint4(pk2(t2.x,t2.y), pk2(t2.z,t2.w), pk2(t3.x,t3.y), pk2(t3.z,t3.w));
    }
    __syncthreads();

    #pragma unroll 2
    for (int ks = 0; ks < KSTEPS; ++ks) {
        const int cur = ks & 1;
        const bool hasNext = (ks + 1) < KSTEPS;
        float4 t0, t1, t2, t3;
        if (hasNext) {                               // hoist next chunk's global loads
            const float4* s4 = (const float4*)srcPtr(ks + 1);
            t0 = s4[0]; t1 = s4[1]; t2 = s4[2]; t3 = s4[3];
        }

        Frag a[2], bfr[4];
        #pragma unroll
        for (int sm = 0; sm < 2; ++sm) {             // A frags from LDS (ds_load_b128)
            int mrow = waveM * 32 + sm * 16 + laneLo;
            const unsigned int* basep = &a_lds[cur][mrow * 20 + laneHi * 4];
            a[sm].q[0] = *(const uint4*)(basep);
            a[sm].q[1] = *(const uint4*)(basep + 8);
        }
        #pragma unroll
        for (int sn = 0; sn < 4; ++sn) {             // B frags from global/L2
            bfr[sn].q[0] = bp[sn][ks * 4];
            bfr[sn].q[1] = bp[sn][ks * 4 + 2];
        }

        #pragma unroll
        for (int sm = 0; sm < 2; ++sm)
            #pragma unroll
            for (int sn = 0; sn < 4; ++sn)
                acc[sm][sn] = __builtin_amdgcn_wmma_f32_16x16x32_bf16(
                    false, a[sm].v, false, bfr[sn].v, (short)0, acc[sm][sn],
                    false, false);

        if (hasNext) {                               // commit next A tile
            unsigned int* dst = &a_lds[(ks + 1) & 1][r * 20 + half * 8];
            ((uint4*)dst)[0] = make_uint4(pk2(t0.x,t0.y), pk2(t0.z,t0.w), pk2(t1.x,t1.y), pk2(t1.z,t1.w));
            ((uint4*)dst)[1] = make_uint4(pk2(t2.x,t2.y), pk2(t2.z,t2.w), pk2(t3.x,t3.y), pk2(t3.z,t3.w));
        }
        __syncthreads();
    }

    // epilogue: bias + tanh + residual, C layout: VGPR rr -> M=rr(+8 upper lanes), N=lane&15
    #pragma unroll
    for (int sm = 0; sm < 2; ++sm) {
        const int mTop = mBase + waveM * 32 + sm * 16 + laneHi * 8;
        #pragma unroll
        for (int sn = 0; sn < 4; ++sn) {
            const int n = nBase + waveN * 64 + sn * 16 + laneLo;
            const float bias = bvec[i * D1 + n];
            #pragma unroll
            for (int rr = 0; rr < 8; ++rr) {
                const int m   = mTop + rr;
                const int idx = (m * N_EL + i) * D1 + n;
                out[idx] = tanhf(acc[sm][sn][rr] + bias) + sh[idx];
            }
        }
    }
}

// ---------------------------------------------------------------------------
// Double stream (memory-bound, K=32): out[b,i,j,d] = tanh(w[i,j,d,:]·h + c) + h[d]
// One wave per (b,i,j) row.
// ---------------------------------------------------------------------------
__global__ void __launch_bounds__(256)
fermi_double(const float* __restrict__ dh, const float* __restrict__ w,
             const float* __restrict__ c, float* __restrict__ out)
{
    __shared__ float hrow[8][D2];
    const int lane = threadIdx.x & 31;
    const int wv   = threadIdx.x >> 5;
    const int rid  = blockIdx.x * 8 + wv;            // (b*N+i)*N + j
    const int ij   = rid & (N_EL * N_EL - 1);
    const float* h = dh + (size_t)rid * D2;

    hrow[wv][lane] = h[lane];
    __syncthreads();

    const float4* w4 = (const float4*)(w + ((size_t)ij * D2 + lane) * D2);
    const float4* h4 = (const float4*)hrow[wv];
    float acc = 0.f;
    #pragma unroll
    for (int e = 0; e < D2 / 4; ++e) {
        float4 a = w4[e], b = h4[e];
        acc += a.x * b.x + a.y * b.y + a.z * b.z + a.w * b.w;
    }
    out[(size_t)rid * D2 + lane] = tanhf(acc + c[ij * D2 + lane]) + hrow[wv][lane];
}

// ---------------------------------------------------------------------------
extern "C" void kernel_launch(void* const* d_in, const int* in_sizes, int n_in,
                              void* d_out, int out_size, void* d_ws, size_t ws_size,
                              hipStream_t stream) {
    const float* sh  = (const float*)d_in[0];   // single_h  (B,N,D1)
    const float* dh  = (const float*)d_in[1];   // double_h  (B,N,N,D2)
    const float* v   = (const float*)d_in[2];   // v_matrices (N,FLEN,D1)
    const float* bv  = (const float*)d_in[3];   // b_vectors (N,D1)
    const float* w   = (const float*)d_in[4];   // w_matrices (N,N,D2,D2)
    const float* c   = (const float*)d_in[5];   // c_vectors (N,N,D2)
    const int*   nup = (const int*)d_in[6];     // n_up scalar

    char* ws = (char*)d_ws;
    float*          gup = (float*)(ws);                     //  4 MB
    float*          gdn = (float*)(ws + (4u  << 20));       //  4 MB
    float*          dgu = (float*)(ws + (8u  << 20));       //  8 MB
    float*          dgd = (float*)(ws + (16u << 20));       //  8 MB
    unsigned short* vt  = (unsigned short*)(ws + (24u << 20)); // ~6.5 MB

    float* outS = (float*)d_out;                            // (B,N,D1)
    float* outD = outS + (size_t)B_WALK * N_EL * D1;        // (B,N,N,D2)

    fermi_pool_g <<<(B_WALK * D1 + 255) / 256, 256, 0, stream>>>(sh, gup, gdn, nup);
    fermi_pool_dg<<<(B_WALK * N_EL * D2 + 255) / 256, 256, 0, stream>>>(dh, dgu, dgd, nup);
    fermi_prep_vt<<<(N_EL * (FLEN / 2) * D1 + 255) / 256, 256, 0, stream>>>(
        v, (unsigned int*)vt);

    fermi_single_gemm<<<dim3(B_WALK / 128, D1 / 128, N_EL), 256, 0, stream>>>(
        sh, gup, gdn, dgu, dgd, vt, bv, outS);

    fermi_double<<<(B_WALK * N_EL * N_EL) / 8, 256, 0, stream>>>(dh, w, c, outD);
}
